// PreFourierAttention_33208687132775
// MI455X (gfx1250) — matrix operand order repaired and checked by reading, not compiled, not run on hardware
//
#include <hip/hip_runtime.h>
#include <hip/hip_bf16.h>

typedef _Float16 v16h __attribute__((ext_vector_type(16)));
typedef _Float16 h8   __attribute__((ext_vector_type(8)));
typedef float    v8f  __attribute__((ext_vector_type(8)));

#define BSTR   (1L << 21)   // B stride in elements (L*H*D = 4096*512)
#define SEGSTR 32768        // segment stride (L_s*H*D)
#define LSTR   512          // l stride (H*D)

// ---------------- Kernel 0: build C = (1/64) * sum_{d=0}^{32} e^{-i w d (t-s)} ----------------
// layout: Cf[t][comp*64+s], comp 0 = Re, comp 1 = Im
__global__ void k_build_c(_Float16* __restrict__ Cf) {
    const float w0 = 6.2831853071795864f / 64.0f;
    for (int idx = threadIdx.x; idx < 64 * 128; idx += blockDim.x) {
        int t = idx >> 7;
        int c = idx & 127;
        int s = c & 63;
        int comp = c >> 6;
        float th = w0 * (float)(t - s);
        float acc = 0.f;
        for (int d = 0; d <= 32; ++d) {
            float ang = th * (float)d;
            acc += (comp == 0) ? cosf(ang) : -sinf(ang);
        }
        Cf[idx] = (_Float16)(acc * (1.0f / 64.0f));
    }
}

// ---------------- Kernel 1: convert K (transposed) and V (shifted) to f16 ----------------
// KcT[bh][l*64+d][n]  (n = key segment 0..62, col 63 zero)
// Vc [bh][n][l*64+d]  (n = value segment n+1, row 63 zero)
__global__ void k_convert(const float* __restrict__ Kin, const float* __restrict__ Vin,
                          _Float16* __restrict__ KcT, _Float16* __restrict__ Vc) {
    __shared__ _Float16 lds[64 * 65];
    int blk = blockIdx.x;
    int bh = blk >> 6;
    int l  = blk & 63;
    int b = bh >> 3, h = bh & 7;
    long base = ((long)b << 21) + (long)l * LSTR + h * 64;

    for (int i = threadIdx.x; i < 4096; i += blockDim.x) {
        int n = i >> 6, d = i & 63;
        float v = (n < 63) ? Kin[base + (long)n * SEGSTR + d] : 0.f;
        lds[n * 65 + d] = (_Float16)v;
    }
    __syncthreads();
    long kout = ((long)bh * 4096 + l * 64) * 64;
    for (int i = threadIdx.x; i < 4096; i += blockDim.x) {
        int d = i >> 6, n = i & 63;
        KcT[kout + (long)d * 64 + n] = lds[n * 65 + d];
    }
    long vout = (long)bh * 262144 + l * 64;
    for (int i = threadIdx.x; i < 4096; i += blockDim.x) {
        int n = i >> 6, d = i & 63;
        float v = (n < 63) ? Vin[base + (long)(n + 1) * SEGSTR + d] : 0.f;
        Vc[vout + (long)n * 4096 + d] = (_Float16)v;
    }
}

// ---------------- Kernel 2: transform Q: Qt[bh][comp*64+m][l*64+s] = sum_t q[t]*C[t][comp*64+s] ----------------
__global__ void k_transform(const float* __restrict__ Qin, const _Float16* __restrict__ Cf,
                            _Float16* __restrict__ Qt) {
    int wid  = blockIdx.x * (blockDim.x >> 5) + (threadIdx.x >> 5);   // 0..32767
    int lane = threadIdx.x & 31;
    int bh = wid >> 8;
    int rt = wid & 255;          // row tile within (b,h): 16 rows = fixed m, l0..l0+15
    int m  = rt >> 2;
    int l0 = (rt & 3) << 4;
    int b = bh >> 3, h = bh & 7;
    int lrow = l0 + (lane & 15);
    int kb = (lane & 16) ? 8 : 0;
    const float* rp = Qin + ((long)b << 21) + (long)m * SEGSTR + (long)lrow * LSTR + h * 64;

    v16h afr[2];
#pragma unroll
    for (int ks = 0; ks < 2; ++ks) {
        int t0 = ks * 32 + kb;
        float4 c0 = *(const float4*)(rp + t0);
        float4 c1 = *(const float4*)(rp + t0 + 4);
        float4 c2 = *(const float4*)(rp + t0 + 16);
        float4 c3 = *(const float4*)(rp + t0 + 20);
        v16h a;
        a[0]=(_Float16)c0.x;  a[1]=(_Float16)c0.y;  a[2]=(_Float16)c0.z;  a[3]=(_Float16)c0.w;
        a[4]=(_Float16)c1.x;  a[5]=(_Float16)c1.y;  a[6]=(_Float16)c1.z;  a[7]=(_Float16)c1.w;
        a[8]=(_Float16)c2.x;  a[9]=(_Float16)c2.y;  a[10]=(_Float16)c2.z; a[11]=(_Float16)c2.w;
        a[12]=(_Float16)c3.x; a[13]=(_Float16)c3.y; a[14]=(_Float16)c3.z; a[15]=(_Float16)c3.w;
        afr[ks] = a;
    }

    _Float16* Qtb = Qt + (long)bh * 524288;
    int lo_ = l0 + ((lane & 16) ? 8 : 0);
#pragma unroll
    for (int ct = 0; ct < 8; ++ct) {
        int n0 = ct * 16;
        v8f acc = {0.f,0.f,0.f,0.f,0.f,0.f,0.f,0.f};
#pragma unroll
        for (int ks = 0; ks < 2; ++ks) {
            int kk = ks * 32 + ((lane & 16) ? 8 : 0);
            v16h bf;
#pragma unroll
            for (int i = 0; i < 8; ++i) { bf[i] = Cf[(kk + i) * 128 + n0 + (lane & 15)]; bf[8 + i] = Cf[(kk + 16 + i) * 128 + n0 + (lane & 15)]; }
            acc = __builtin_amdgcn_wmma_f32_16x16x32_f16(false, afr[ks], false, bf, (short)0, acc, false, false);
        }
        int comp = n0 >> 6;
        int s = (n0 & 63) + (lane & 15);
        _Float16* orow = Qtb + (long)(comp * 64 + m) * 4096;
#pragma unroll
        for (int r = 0; r < 8; ++r)
            orow[(lo_ + r) * 64 + s] = (_Float16)acc[r];
    }
}

// ---------------- Kernel 3: scores S[bh][128][64] = Qt[128][4096] @ KcT[4096][64] ----------------
__global__ void k_scores(const _Float16* __restrict__ Qt, const _Float16* __restrict__ KcT,
                         float* __restrict__ S) {
    int bh = blockIdx.x;
    int w = threadIdx.x >> 5;
    int lane = threadIdx.x & 31;
    int r0 = w * 16;
    const _Float16* Qtb = Qt + (long)bh * 524288;
    const _Float16* Kb  = KcT + (long)bh * 262144;
    int arow = r0 + (lane & 15);
    int kb = (lane & 16) ? 8 : 0;
    const _Float16* ap = Qtb + (long)arow * 4096 + kb;

    v8f acc[4];
    v8f z = {0.f,0.f,0.f,0.f,0.f,0.f,0.f,0.f};
    acc[0] = z; acc[1] = z; acc[2] = z; acc[3] = z;

    for (int k0 = 0; k0 < 4096; k0 += 32) {
        h8 lo = *(const h8*)(ap + k0);
        h8 hi = *(const h8*)(ap + k0 + 16);
        v16h a;
#pragma unroll
        for (int i = 0; i < 8; ++i) { a[i] = lo[i]; a[8 + i] = hi[i]; }
        const _Float16* kp = Kb + (long)(k0 + kb) * 64 + (lane & 15);
#pragma unroll
        for (int ct = 0; ct < 4; ++ct) {
            v16h bf;
#pragma unroll
            for (int i = 0; i < 8; ++i) { bf[i] = kp[i * 64 + ct * 16]; bf[8 + i] = kp[(16 + i) * 64 + ct * 16]; }
            acc[ct] = __builtin_amdgcn_wmma_f32_16x16x32_f16(false, a, false, bf, (short)0, acc[ct], false, false);
        }
    }
    float* Sb = S + (long)bh * 8192;
    int rbase = r0 + ((lane & 16) ? 8 : 0);
    int nl = lane & 15;
#pragma unroll
    for (int ct = 0; ct < 4; ++ct)
#pragma unroll
        for (int r = 0; r < 8; ++r)
            Sb[(rbase + r) * 64 + ct * 16 + nl] = acc[ct][r];
}

// ---------------- Kernel 4: softmax of |S|/512 over 63 columns -> A f16 (col 63 = 0) ----------------
__global__ void k_softmax(const float* __restrict__ S, _Float16* __restrict__ Amat) {
    int bh = blockIdx.x;
    int m = threadIdx.x;  // 0..63
    const float* re = S + (long)bh * 8192 + m * 64;
    const float* im = re + 4096;
    const float sc = 1.0f / 512.0f;
    float mx = -1e30f;
    for (int n = 0; n < 63; ++n) {
        float a = re[n], b = im[n];
        mx = fmaxf(mx, sqrtf(a * a + b * b) * sc);
    }
    _Float16* arow = Amat + (long)bh * 4096 + m * 64;
    float sum = 0.f;
    for (int n = 0; n < 63; ++n) {
        float a = re[n], b = im[n];
        float e = __expf(sqrtf(a * a + b * b) * sc - mx);
        sum += e;
        arow[n] = (_Float16)e;
    }
    float inv = 1.0f / sum;
    for (int n = 0; n < 63; ++n)
        arow[n] = (_Float16)((float)arow[n] * inv);
    arow[63] = (_Float16)0.f;
}

// ---------------- Kernel 5: V-mix + roll: out[b][(m+1)%64][l][h][d] = (A @ Vc)[m][l*64+d] ----------------
__global__ void k_vmix(const _Float16* __restrict__ Amat, const _Float16* __restrict__ Vc,
                       float* __restrict__ Out) {
    int wid  = blockIdx.x * (blockDim.x >> 5) + (threadIdx.x >> 5);  // 0..4095
    int lane = threadIdx.x & 31;
    int bh  = wid >> 5;
    int rem = wid & 31;
    int mt = rem >> 3;     // row tile 0..3
    int cc = rem & 7;      // 512-wide column chunk
    int b = bh >> 3, h = bh & 7;
    int kb = (lane & 16) ? 8 : 0;

    const _Float16* arow = Amat + (long)bh * 4096 + (mt * 16 + (lane & 15)) * 64 + kb;
    v16h afr[2];
#pragma unroll
    for (int ks = 0; ks < 2; ++ks) {
        h8 lo = *(const h8*)(arow + ks * 32);
        h8 hi = *(const h8*)(arow + ks * 32 + 16);
        v16h a;
#pragma unroll
        for (int i = 0; i < 8; ++i) { a[i] = lo[i]; a[8 + i] = hi[i]; }
        afr[ks] = a;
    }

    const _Float16* Vb = Vc + (long)bh * 262144;
    float* ob = Out + ((long)b << 21) + h * 64;
    int mbase = mt * 16 + ((lane & 16) ? 8 : 0);

    for (int ct = 0; ct < 32; ++ct) {
        int col0 = cc * 512 + ct * 16;
        v8f acc = {0.f,0.f,0.f,0.f,0.f,0.f,0.f,0.f};
#pragma unroll
        for (int ks = 0; ks < 2; ++ks) {
            int kk = ks * 32 + kb;
            v16h bf;
#pragma unroll
            for (int i = 0; i < 8; ++i) { bf[i] = Vb[(long)(kk + i) * 4096 + col0 + (lane & 15)]; bf[8 + i] = Vb[(long)(kk + 16 + i) * 4096 + col0 + (lane & 15)]; }
            acc = __builtin_amdgcn_wmma_f32_16x16x32_f16(false, afr[ks], false, bf, (short)0, acc, false, false);
        }
        int l = col0 >> 6;
        int d = (col0 & 63) + (lane & 15);
#pragma unroll
        for (int r = 0; r < 8; ++r) {
            int mm = mbase + r;
            int seg = (mm + 1) & 63;
            ob[(long)seg * SEGSTR + (long)l * LSTR + d] = acc[r];
        }
    }
}

extern "C" void kernel_launch(void* const* d_in, const int* in_sizes, int n_in,
                              void* d_out, int out_size, void* d_ws, size_t ws_size,
                              hipStream_t stream) {
    (void)in_sizes; (void)n_in; (void)out_size; (void)ws_size;
    const float* q = (const float*)d_in[0];
    const float* k = (const float*)d_in[1];
    const float* v = (const float*)d_in[2];
    float* out = (float*)d_out;
    char* ws = (char*)d_ws;

    _Float16* Cf   = (_Float16*)(ws);                                   //   16 KB
    _Float16* Qt   = (_Float16*)(ws + 16384);                           //  128 MB
    _Float16* KcT  = (_Float16*)(ws + 16384 + 134217728);               //   64 MB
    _Float16* Vc   = (_Float16*)(ws + 16384 + 201326592);               //   64 MB
    float*    S    = (float*)   (ws + 16384 + 268435456);               //    4 MB
    _Float16* Amat = (_Float16*)(ws + 16384 + 268435456 + 4194304);     //    1 MB

    k_build_c <<<1,    256, 0, stream>>>(Cf);
    k_convert <<<8192, 256, 0, stream>>>(k, v, KcT, Vc);
    k_transform<<<4096, 256, 0, stream>>>(q, Cf, Qt);
    k_scores  <<<128,  256, 0, stream>>>(Qt, KcT, S);
    k_softmax <<<128,   64, 0, stream>>>(S, Amat);
    k_vmix    <<<512,  256, 0, stream>>>(Amat, Vc, out);
}
